// MultiHeadSelfAttention_31808527794756
// MI455X (gfx1250) — matrix operand order repaired and checked by reading, not compiled, not run on hardware
//
#include <hip/hip_runtime.h>

// ---------------------------------------------------------------------------
// MHA with RoPE for MI455X (gfx1250): bf16 WMMA everywhere, f32 accumulate.
// B=4, S=2048, D=1024, H=16, DH=64.
// ---------------------------------------------------------------------------

#define CB  4
#define CS  2048
#define CD  1024
#define CH  16
#define CDH 64

typedef __attribute__((ext_vector_type(16))) __bf16 v16bf;
typedef __attribute__((ext_vector_type(8)))  float  v8f;

union ABfrag { v16bf bf; unsigned int u[8]; };

__device__ __forceinline__ unsigned short f2bf(float f) {
  unsigned int u = __float_as_uint(f);
  u += 0x7FFFu + ((u >> 16) & 1u);           // round-to-nearest-even
  return (unsigned short)(u >> 16);
}

// A-matrix 16x32 bf16 fragment from row-major [16][>=32] source (stride ld).
// ISA layout: lanes 0-15 -> K 0..7 & 16..23 ; lanes 16-31 -> K 8..15 & 24..31.
__device__ __forceinline__ void load_a16x32(const unsigned short* src,
                                            int ld, ABfrag& a, int lane) {
  int r  = lane & 15;
  int kb = (lane & 16) ? 8 : 0;
  const unsigned short* p = src + r * ld + kb;
#pragma unroll
  for (int j = 0; j < 4; ++j) {
    a.u[j]     = *(const unsigned int*)(p + 2 * j);
    a.u[j + 4] = *(const unsigned int*)(p + 16 + 2 * j);
  }
}

// B-matrix 32x16 bf16 fragment from B^T stored row-major [n][k] (stride ld).
// ISA layout: lanes 0-15 -> K 0..7 & 16..23 ; lanes 16-31 -> K 8..15 & 24..31 ; 2 bf16/dword.
__device__ __forceinline__ void load_bt32x16(const unsigned short* src,
                                             int ld, ABfrag& b, int lane) {
  int n  = lane & 15;
  int kb = (lane & 16) ? 8 : 0;
  const unsigned short* p = src + n * ld + kb;
#pragma unroll
  for (int j = 0; j < 4; ++j) {
    b.u[j]     = *(const unsigned int*)(p + 2 * j);
    b.u[j + 4] = *(const unsigned int*)(p + 16 + 2 * j);
  }
}

__device__ __forceinline__ v8f wmma_bf16(const ABfrag& a, const ABfrag& b, v8f c) {
  return __builtin_amdgcn_wmma_f32_16x16x32_bf16(false, a.bf, false, b.bf,
                                                 (short)0, c, false, false);
}

__device__ __forceinline__ float rmax16(float v) {
#pragma unroll
  for (int m = 1; m < 16; m <<= 1) v = fmaxf(v, __shfl_xor(v, m, 32));
  return v;
}
__device__ __forceinline__ float rsum16(float v) {
#pragma unroll
  for (int m = 1; m < 16; m <<= 1) v += __shfl_xor(v, m, 32);
  return v;
}

// ---------------------------------------------------------------------------
// fp32 -> bf16 conversion, 4 elements / thread
// ---------------------------------------------------------------------------
__global__ void cvt_bf16x4(const float* __restrict__ src,
                           unsigned short* __restrict__ dst, int n) {
  int i = (blockIdx.x * blockDim.x + threadIdx.x) * 4;
  if (i + 3 < n) {
    float4 f = *(const float4*)(src + i);
    unsigned int u0 = (unsigned int)f2bf(f.x) | ((unsigned int)f2bf(f.y) << 16);
    unsigned int u1 = (unsigned int)f2bf(f.z) | ((unsigned int)f2bf(f.w) << 16);
    *(unsigned int*)(dst + i)     = u0;
    *(unsigned int*)(dst + i + 2) = u1;
  }
}

// ---------------------------------------------------------------------------
// GEMM  Y = X * W^T   (X: [M,1024] bf16 row-major, W: [1024,1024] bf16, row=n)
// One wave -> 16(M) x 64(N) strip, 4 waves/block -> 64x64 block tile.
// MODE 0: RoPE epilogue, store bf16 [B,H,S,DH]   (Q and K)
// MODE 1: store bf16 transposed [B,H,DH,S]       (V)
// MODE 2: store f32 [B,S,D]                      (output projection)
// ---------------------------------------------------------------------------
template <int MODE>
__global__ __launch_bounds__(128) void proj_gemm(
    const unsigned short* __restrict__ X,
    const unsigned short* __restrict__ W,
    const int* __restrict__ tpos,
    unsigned short* __restrict__ dstb,
    float* __restrict__ dstf) {
  int lane = threadIdx.x & 31;
  int wv   = threadIdx.x >> 5;
  int m0   = blockIdx.y * 64 + wv * 16;
  int n0   = blockIdx.x * 64;

  v8f acc[4] = {};
  for (int k = 0; k < CD; k += 32) {
    ABfrag a;
    load_a16x32(X + (size_t)m0 * CD + k, CD, a, lane);
#pragma unroll
    for (int t = 0; t < 4; ++t) {
      ABfrag b;
      load_bt32x16(W + (size_t)(n0 + t * 16) * CD + k, CD, b, lane);
      acc[t] = wmma_bf16(a, b, acc[t]);
    }
  }

  int half8 = (lane & 16) ? 8 : 0;
  int col   = lane & 15;

  if (MODE == 2) {
#pragma unroll
    for (int t = 0; t < 4; ++t)
#pragma unroll
      for (int r = 0; r < 8; ++r) {
        int row = m0 + r + half8;
        dstf[(size_t)row * CD + n0 + t * 16 + col] = acc[t][r];
      }
  } else if (MODE == 1) {                      // V, transposed [BH, DH, S]
    int h = blockIdx.x;                        // 64-wide N tile == one head
#pragma unroll
    for (int t = 0; t < 4; ++t)
#pragma unroll
      for (int r = 0; r < 8; ++r) {
        int row  = m0 + r + half8;
        int bidx = row >> 11, s = row & (CS - 1);
        int dh   = t * 16 + col;
        dstb[((size_t)(bidx * CH + h) * CDH + dh) * CS + s] = f2bf(acc[t][r]);
      }
  } else {                                     // Q/K with fused RoPE
    int h = blockIdx.x;
#pragma unroll
    for (int t = 0; t < 4; ++t)
#pragma unroll
      for (int r = 0; r < 8; ++r) {
        int row  = m0 + r + half8;
        int bidx = row >> 11, s = row & (CS - 1);
        int dh   = t * 16 + col;
        float v  = acc[t][r];
        float p  = __shfl_xor(v, 1, 32);       // partner of the (even,odd) pair
        float pos = (float)tpos[s];
        // inv_freq = theta^(-(2j)/DH) = exp(-(dh&~1) * ln(10000)/64)
        float ang = pos * __expf((float)(dh & ~1) * -0.14391156831212787f);
        float sn, cn;
        __sincosf(ang, &sn, &cn);
        float out = (dh & 1) ? (p * sn + v * cn) : (v * cn - p * sn);
        dstb[((size_t)(bidx * CH + h) * CS + s) * CDH + dh] = f2bf(out);
      }
  }
}

// ---------------------------------------------------------------------------
// Causal flash attention: one wave per (head, 16-query tile), 32-key blocks.
// Q,K: [BH,S,DH] bf16 (RoPE applied) ; Vt: [BH,DH,S] bf16 ; O: [B,S,D] bf16.
// ---------------------------------------------------------------------------
__global__ __launch_bounds__(128) void attn_kernel(
    const unsigned short* __restrict__ Q,
    const unsigned short* __restrict__ K,
    const unsigned short* __restrict__ Vt,
    unsigned short* __restrict__ O) {
  __shared__ unsigned short plds[4][16][32];   // P staging, 1 KB per wave

  int lane = threadIdx.x & 31;
  int w    = threadIdx.x >> 5;
  int qt   = blockIdx.x * 4 + w;
  int q0   = qt * 16;
  int bh   = blockIdx.y;

  const unsigned short* Qp = Q  + (size_t)bh * CS * CDH;
  const unsigned short* Kp = K  + (size_t)bh * CS * CDH;
  const unsigned short* Vp = Vt + (size_t)bh * CDH * CS;

  ABfrag qa0, qa1;                             // Q tile, K-dim = DH split 2x32
  load_a16x32(Qp + (size_t)q0 * CDH,      CDH, qa0, lane);
  load_a16x32(Qp + (size_t)q0 * CDH + 32, CDH, qa1, lane);

  v8f o[4] = {};
  float mrun[8], lrun[8];
#pragma unroll
  for (int r = 0; r < 8; ++r) { mrun[r] = -3.0e38f; lrun[r] = 0.0f; }

  int half8 = (lane & 16) ? 8 : 0;
  int col   = lane & 15;

  for (int k0 = 0; k0 < q0 + 16; k0 += 32) {
    // ---- scores: two 16-key C tiles, K-dim = 64 (2 WMMAs each) ----
    v8f sv[2];
#pragma unroll
    for (int t = 0; t < 2; ++t) {
      int kc = k0 + t * 16;
      v8f z = {};
      ABfrag b;
      load_bt32x16(Kp + (size_t)kc * CDH,      CDH, b, lane);
      z = wmma_bf16(qa0, b, z);
      load_bt32x16(Kp + (size_t)kc * CDH + 32, CDH, b, lane);
      z = wmma_bf16(qa1, b, z);
      sv[t] = z;
    }
    // ---- online softmax over this 32-key block ----
    float alpha[8];
#pragma unroll
    for (int r = 0; r < 8; ++r) {
      int qrow = q0 + r + half8;
      float s0 = sv[0][r] * 0.125f;            // 1/sqrt(64)
      float s1 = sv[1][r] * 0.125f;
      if (k0 + col > qrow)      s0 = -3.0e38f; // causal mask
      if (k0 + 16 + col > qrow) s1 = -3.0e38f;
      float tm = rmax16(fmaxf(s0, s1));
      float mn = fmaxf(mrun[r], tm);
      float al = __expf(mrun[r] - mn);
      mrun[r]  = mn;
      float p0 = __expf(s0 - mn);
      float p1 = __expf(s1 - mn);
      lrun[r]  = lrun[r] * al + rsum16(p0 + p1);
      alpha[r] = al;
      int row = r + half8;
      plds[w][row][col]      = f2bf(p0);       // C-layout -> row-major in LDS
      plds[w][row][col + 16] = f2bf(p1);
    }
#pragma unroll
    for (int t = 0; t < 4; ++t)
#pragma unroll
      for (int r = 0; r < 8; ++r) o[t][r] *= alpha[r];

    asm volatile("s_wait_dscnt 0" ::: "memory");   // LDS store -> load fence

    ABfrag pa;                                 // reload P as A fragment
    load_a16x32(&plds[w][0][0], 32, pa, lane);
#pragma unroll
    for (int t = 0; t < 4; ++t) {              // O += P * V (4 DH tiles)
      ABfrag vb;
      load_bt32x16(Vp + (size_t)(t * 16) * CS + k0, CS, vb, lane);
      o[t] = wmma_bf16(pa, vb, o[t]);
    }
  }

  // ---- normalize + store to [B,S,D] bf16 ----
  int b = bh >> 4, h = bh & 15;
#pragma unroll
  for (int r = 0; r < 8; ++r) {
    float inv = 1.0f / lrun[r];
    int q = q0 + r + half8;
#pragma unroll
    for (int t = 0; t < 4; ++t) {
      int dh = t * 16 + col;
      O[(size_t)(b * CS + q) * CD + h * CDH + dh] = f2bf(o[t][r] * inv);
    }
  }
}

// ---------------------------------------------------------------------------
extern "C" void kernel_launch(void* const* d_in, const int* in_sizes, int n_in,
                              void* d_out, int out_size, void* d_ws, size_t ws_size,
                              hipStream_t stream) {
  (void)in_sizes; (void)n_in; (void)out_size; (void)ws_size;
  const float* x    = (const float*)d_in[0];
  const int*   tpos = (const int*)d_in[1];
  const float* wq   = (const float*)d_in[2];
  const float* wk   = (const float*)d_in[3];
  const float* wv   = (const float*)d_in[4];
  const float* wo   = (const float*)d_in[5];
  float* out = (float*)d_out;

  const size_t nX = (size_t)CB * CS * CD;      // 8,388,608 elements
  const size_t nW = (size_t)CD * CD;           // 1,048,576 elements

  char* ws = (char*)d_ws;                      // ~88 MB total
  unsigned short* Xb  = (unsigned short*)ws;  ws += nX * 2;
  unsigned short* Wqb = (unsigned short*)ws;  ws += nW * 2;
  unsigned short* Wkb = (unsigned short*)ws;  ws += nW * 2;
  unsigned short* Wvb = (unsigned short*)ws;  ws += nW * 2;
  unsigned short* Wob = (unsigned short*)ws;  ws += nW * 2;
  unsigned short* Qr  = (unsigned short*)ws;  ws += nX * 2;   // [BH,S,DH]
  unsigned short* Kr  = (unsigned short*)ws;  ws += nX * 2;   // [BH,S,DH]
  unsigned short* Vt  = (unsigned short*)ws;  ws += nX * 2;   // [BH,DH,S]
  unsigned short* At  = (unsigned short*)ws;  ws += nX * 2;   // [B,S,D]

  cvt_bf16x4<<<dim3((unsigned)(nX / 4 / 256)), 256, 0, stream>>>(x,  Xb,  (int)nX);
  cvt_bf16x4<<<dim3((unsigned)(nW / 4 / 256)), 256, 0, stream>>>(wq, Wqb, (int)nW);
  cvt_bf16x4<<<dim3((unsigned)(nW / 4 / 256)), 256, 0, stream>>>(wk, Wkb, (int)nW);
  cvt_bf16x4<<<dim3((unsigned)(nW / 4 / 256)), 256, 0, stream>>>(wv, Wvb, (int)nW);
  cvt_bf16x4<<<dim3((unsigned)(nW / 4 / 256)), 256, 0, stream>>>(wo, Wob, (int)nW);

  dim3 gp(CD / 64, (CB * CS) / 64);            // (16, 128), 128 thr/block
  proj_gemm<0><<<gp, 128, 0, stream>>>(Xb, Wqb, tpos, Qr, nullptr);
  proj_gemm<0><<<gp, 128, 0, stream>>>(Xb, Wkb, tpos, Kr, nullptr);
  proj_gemm<1><<<gp, 128, 0, stream>>>(Xb, Wvb, tpos, Vt, nullptr);

  attn_kernel<<<dim3(CS / 64, CB * CH), 128, 0, stream>>>(Qr, Kr, Vt, At);

  proj_gemm<2><<<gp, 128, 0, stream>>>(At, Wob, tpos, nullptr, out);
}
